// GAT_40372692582699
// MI455X (gfx1250) — matrix operand order repaired and checked by reading, not compiled, not run on hardware
//
#include <hip/hip_runtime.h>
#include <cmath>

#define N_NODES 20000
#define N_EDGES 320000
#define D_IN    512
#define HID     128
#define HEADS   8
#define C_OUT   64
#define SLOPE   0.01f

typedef unsigned short u16;
typedef __attribute__((ext_vector_type(16))) __bf16 v16bf;
typedef __attribute__((ext_vector_type(8)))  float  v8f;
typedef __attribute__((ext_vector_type(8)))  u16    us8;

// round-to-nearest-even fp32 -> bf16 bits
__device__ __forceinline__ u16 bf16_rne(float x) {
  unsigned u = __builtin_bit_cast(unsigned, x);
  unsigned r = u + 0x7FFFu + ((u >> 16) & 1u);
  return (u16)(r >> 16);
}
__device__ __forceinline__ float bf16_to_f32(u16 b) {
  return __builtin_bit_cast(float, (unsigned)b << 16);
}
__device__ __forceinline__ v16bf mkfrag(us8 lo8, us8 hi8) {
  struct P { us8 x, y; } p{lo8, hi8};
  return __builtin_bit_cast(v16bf, p);
}

// ---------------------------------------------------------------------------
// Elementwise fp32 -> (bf16 hi, bf16 lo) split, done ONCE per tensor.
// ---------------------------------------------------------------------------
__global__ __launch_bounds__(256)
void split_bf16(const float* __restrict__ in, u16* __restrict__ hi,
                u16* __restrict__ lo, long n) {
  long i = (long)blockIdx.x * 256 + threadIdx.x;
  if (i >= n) return;
  float x = in[i];
  u16 hb = bf16_rne(x);
  hi[i] = hb;
  lo[i] = bf16_rne(x - bf16_to_f32(hb));
}

// ---------------------------------------------------------------------------
// GEMM with bias from pre-split bf16 planes. bf16x3: Ah*Bh + Ah*Bl + Al*Bh.
// grid: (ceil(nrows/128), dh/16, heads); block: 256 = 8 waves, each wave
// owns a 16x16 D tile. din % 32 == 0, dh % 16 == 0.
// Inner loop: 4 global b128 (A), 4 ds b128 (B), 3 WMMAs, 2 prefetches.
// ---------------------------------------------------------------------------
#define LWS 40   // padded LDS row stride (ushorts) per output column
__global__ __launch_bounds__(256)
void wmma_gemm_bias(const u16* __restrict__ Xhi, const u16* __restrict__ Xlo,
                    const u16* __restrict__ Whi, const u16* __restrict__ Wlo,
                    const float* __restrict__ Ball, float* __restrict__ OUT,
                    int nrows, int din, int dh) {
  const int h        = blockIdx.z;
  const int heads_ld = gridDim.z * dh;                 // output row stride
  const u16* WH = Whi + (size_t)h * din * dh;          // [din, dh] row-major
  const u16* WL = Wlo + (size_t)h * din * dh;
  const float* bs = Ball + (size_t)h * dh;
  const int n0   = blockIdx.y * 16;
  const int wave = threadIdx.x >> 5;
  const int lane = threadIdx.x & 31;
  const int hl   = lane >> 4;        // half: lanes 0-15 vs 16-31
  const int l16  = lane & 15;
  const int m0   = blockIdx.x * 128 + wave * 16;

  // B tiles staged in fragment order: [col][K], padded stride
  __shared__ __align__(32) u16 lwh[16 * LWS];
  __shared__ __align__(32) u16 lwl[16 * LWS];

  int arow = m0 + l16;
  if (arow >= nrows) arow = nrows - 1;                 // clamp loads, guard stores
  const u16* xh = Xhi + (size_t)arow * din;
  const u16* xl = Xlo + (size_t)arow * din;

  // staging indices: each thread stages exactly 2 elements per K-step
  const int t0 = threadIdx.x;          // 0..255
  const int t1 = threadIdx.x + 256;    // 256..511
  const int k0a = t0 >> 4, c0a = t0 & 15;
  const int k1a = t1 >> 4, c1a = t1 & 15;

  v8f acc = {};
  for (int k0 = 0; k0 < din; k0 += 32) {
    __syncthreads();
    // stage W[k0:k0+32, n0:n0+16] transposed into LDS (512 elems, 2/thread)
    {
      size_t g0 = (size_t)(k0 + k0a) * dh + (n0 + c0a);
      size_t g1 = (size_t)(k0 + k1a) * dh + (n0 + c1a);
      lwh[c0a * LWS + k0a] = WH[g0];
      lwl[c0a * LWS + k0a] = WL[g0];
      lwh[c1a * LWS + k1a] = WH[g1];
      lwl[c1a * LWS + k1a] = WL[g1];
    }
    // prefetch next K-step of the A stream (global_prefetch_b8)
    if (k0 + 32 < din) {
      __builtin_prefetch(xh + k0 + 32 + 8 * hl, 0, 3);
      __builtin_prefetch(xl + k0 + 32 + 8 * hl, 0, 3);
    }
    __syncthreads();

    // A fragment (ISA 16-bit A layout): halves 0..7 = K k0+8*hl..+7,
    // halves 8..15 = K k0+16+8*hl..+7  -> two contiguous 16B loads per plane
    v16bf Ah = mkfrag(*(const us8*)(xh + k0 + 8 * hl),
                      *(const us8*)(xh + k0 + 16 + 8 * hl));
    v16bf Al = mkfrag(*(const us8*)(xl + k0 + 8 * hl),
                      *(const us8*)(xl + k0 + 16 + 8 * hl));
    // B fragment: lane = col l16, halves 0..7 = K 8*hl..+7, halves 8..15 = K 16+8*hl..+7
    const u16* bh = &lwh[l16 * LWS + 8 * hl];
    const u16* bl = &lwl[l16 * LWS + 8 * hl];
    v16bf Bh = mkfrag(*(const us8*)bh, *(const us8*)(bh + 16));
    v16bf Bl = mkfrag(*(const us8*)bl, *(const us8*)(bl + 16));

    acc = __builtin_amdgcn_wmma_f32_16x16x32_bf16(false, Ah, false, Bh, (short)0, acc, false, false);
    acc = __builtin_amdgcn_wmma_f32_16x16x32_bf16(false, Ah, false, Bl, (short)0, acc, false, false);
    acc = __builtin_amdgcn_wmma_f32_16x16x32_bf16(false, Al, false, Bh, (short)0, acc, false, false);
  }

  // D layout: lane l16 = col, VGPR r -> row m0 + r + 8*hl
  float bb = bs[n0 + l16];
#pragma unroll
  for (int r = 0; r < 8; ++r) {
    int row = m0 + r + 8 * hl;
    if (row < nrows)
      OUT[(size_t)row * heads_ld + h * dh + n0 + l16] = acc[r] + bb;
  }
}

// ---------------------------------------------------------------------------
// a1[n,h] = ft[n,h,:].al[h] + alb[h] ; a2 likewise. Wave per (node, head).
// ---------------------------------------------------------------------------
__global__ __launch_bounds__(256)
void node_logits(const float* __restrict__ ft,
                 const float* __restrict__ al, const float* __restrict__ alb,
                 const float* __restrict__ ar, const float* __restrict__ arb,
                 float* __restrict__ a1, float* __restrict__ a2,
                 int nrows, int dh, int heads) {
  int wid  = blockIdx.x * 8 + (threadIdx.x >> 5);
  int lane = threadIdx.x & 31;
  if (wid >= nrows * heads) return;
  int n = wid / heads, h = wid % heads;
  const float* f   = ft + (size_t)n * heads * dh + h * dh;
  const float* alh = al + h * dh;
  const float* arh = ar + h * dh;
  float s1 = 0.f, s2 = 0.f;
  for (int i = lane; i < dh; i += 32) {
    float v = f[i];
    s1 += v * alh[i];
    s2 += v * arh[i];
  }
#pragma unroll
  for (int off = 16; off; off >>= 1) {
    s1 += __shfl_xor(s1, off, 32);
    s2 += __shfl_xor(s2, off, 32);
  }
  if (lane == 0) { a1[wid] = s1 + alb[h]; a2[wid] = s2 + arb[h]; }
}

__device__ __forceinline__ unsigned f32_order_enc(float s) {
  unsigned u = __builtin_bit_cast(unsigned, s);
  return (u & 0x80000000u) ? ~u : (u | 0x80000000u);
}
__device__ __forceinline__ float f32_order_dec(unsigned e) {
  unsigned u = (e & 0x80000000u) ? (e ^ 0x80000000u) : ~e;
  return __builtin_bit_cast(float, u);
}

// segment max of leaky-relu edge scores over dst
__global__ __launch_bounds__(256)
void edge_max(const int* __restrict__ src, const int* __restrict__ dst,
              const float* __restrict__ a1, const float* __restrict__ a2,
              unsigned* __restrict__ smax, int nE, int heads) {
  int idx = blockIdx.x * blockDim.x + threadIdx.x;
  if (idx >= nE * heads) return;
  int e = idx / heads, h = idx % heads;
  float s = a1[dst[e] * heads + h] + a2[src[e] * heads + h];
  s = s > 0.f ? s : SLOPE * s;
  atomicMax(&smax[dst[e] * heads + h], f32_order_enc(s));
}

// denom[dst,h] += e ; accum[dst,h,:] += e * ft[src,h,:]   (wave per edge-head)
__global__ __launch_bounds__(256)
void edge_agg(const int* __restrict__ src, const int* __restrict__ dst,
              const float* __restrict__ a1, const float* __restrict__ a2,
              const unsigned* __restrict__ smax, const float* __restrict__ ft,
              float* __restrict__ denom, float* __restrict__ accum,
              int nE, int heads, int dh) {
  int wid  = blockIdx.x * 8 + (threadIdx.x >> 5);
  int lane = threadIdx.x & 31;
  if (wid >= nE * heads) return;
  int e = wid / heads, h = wid % heads;
  int d = dst[e], sN = src[e];
  const float* fs = ft + (size_t)sN * heads * dh + h * dh;
  __builtin_prefetch(fs + lane, 0, 3);
  float s = a1[d * heads + h] + a2[sN * heads + h];
  s = s > 0.f ? s : SLOPE * s;
  float w = __expf(s - f32_order_dec(smax[d * heads + h]));
  if (lane == 0) atomicAdd(&denom[d * heads + h], w);
  float* ad = accum + (size_t)d * heads * dh + h * dh;
  for (int i = lane; i < dh; i += 32) atomicAdd(&ad[i], w * fs[i]);
}

// out = elu(accum / denom)
__global__ __launch_bounds__(256)
void elu_norm(const float* __restrict__ accum, const float* __restrict__ denom,
              float* __restrict__ out, int nrows, int heads, int dh) {
  int idx = blockIdx.x * blockDim.x + threadIdx.x;
  int total = nrows * heads * dh;
  if (idx >= total) return;
  int n = idx / (heads * dh);
  int h = (idx % (heads * dh)) / dh;
  float v = accum[idx] / denom[n * heads + h];
  out[idx] = v > 0.f ? v : (__expf(v) - 1.f);
}

// ---------------------------------------------------------------------------
static void run_layer(const float* x, int din,
                      const float* W, const float* b,
                      const float* al, const float* alb,
                      const float* ar, const float* arb,
                      int heads, int dh,
                      const int* src, const int* dst,
                      u16* xhi, u16* xlo, u16* whi, u16* wlo,
                      float* ft, float* a1, float* a2, unsigned* smax,
                      float* denom, float* accum, float* out,
                      hipStream_t stream) {
  const int HD = heads * dh;
  hipMemsetAsync(smax,  0, (size_t)N_NODES * heads * sizeof(unsigned), stream);
  hipMemsetAsync(denom, 0, (size_t)N_NODES * heads * sizeof(float), stream);
  hipMemsetAsync(accum, 0, (size_t)N_NODES * HD * sizeof(float), stream);

  // pre-split X and W into bf16 hi/lo planes (once per layer)
  long nX = (long)N_NODES * din;
  long nW = (long)heads * din * dh;
  split_bf16<<<(unsigned)((nX + 255) / 256), 256, 0, stream>>>(x, xhi, xlo, nX);
  split_bf16<<<(unsigned)((nW + 255) / 256), 256, 0, stream>>>(W, whi, wlo, nW);

  dim3 g((N_NODES + 127) / 128, dh / 16, heads);
  wmma_gemm_bias<<<g, 256, 0, stream>>>(xhi, xlo, whi, wlo, b, ft,
                                        N_NODES, din, dh);

  int totalNH = N_NODES * heads;
  node_logits<<<(totalNH + 7) / 8, 256, 0, stream>>>(ft, al, alb, ar, arb, a1, a2,
                                                     N_NODES, dh, heads);
  int totalEH = N_EDGES * heads;
  edge_max<<<(totalEH + 255) / 256, 256, 0, stream>>>(src, dst, a1, a2, smax,
                                                      N_EDGES, heads);
  edge_agg<<<(totalEH + 7) / 8, 256, 0, stream>>>(src, dst, a1, a2, smax, ft,
                                                  denom, accum, N_EDGES, heads, dh);
  int totalNC = N_NODES * HD;
  elu_norm<<<(totalNC + 255) / 256, 256, 0, stream>>>(accum, denom, out,
                                                      N_NODES, heads, dh);
}

extern "C" void kernel_launch(void* const* d_in, const int* in_sizes, int n_in,
                              void* d_out, int out_size, void* d_ws, size_t ws_size,
                              hipStream_t stream) {
  const float* features = (const float*)d_in[0];
  const int*   src      = (const int*)d_in[1];
  const int*   dst      = (const int*)d_in[2];
  const float* W0  = (const float*)d_in[3];
  const float* b0  = (const float*)d_in[4];
  const float* al0 = (const float*)d_in[5];
  const float* alb0= (const float*)d_in[6];
  const float* ar0 = (const float*)d_in[7];
  const float* arb0= (const float*)d_in[8];
  const float* W1  = (const float*)d_in[9];
  const float* b1  = (const float*)d_in[10];
  const float* al1 = (const float*)d_in[11];
  const float* alb1= (const float*)d_in[12];
  const float* ar1 = (const float*)d_in[13];
  const float* arb1= (const float*)d_in[14];
  const float* Wf  = (const float*)d_in[15];
  const float* bf_ = (const float*)d_in[16];
  const float* alf = (const float*)d_in[17];
  const float* albf= (const float*)d_in[18];
  const float* arf = (const float*)d_in[19];
  const float* arbf= (const float*)d_in[20];

  // workspace partition
  const size_t HDMAX = (size_t)HEADS * HID;               // 1024
  char* w = (char*)d_ws;
  float* ft    = (float*)w;  w += (size_t)N_NODES * HDMAX * sizeof(float);
  float* accum = (float*)w;  w += (size_t)N_NODES * HDMAX * sizeof(float);
  float* xbuf  = (float*)w;  w += (size_t)N_NODES * HDMAX * sizeof(float);
  u16*  xhi    = (u16*)w;    w += (size_t)N_NODES * HDMAX * sizeof(u16);
  u16*  xlo    = (u16*)w;    w += (size_t)N_NODES * HDMAX * sizeof(u16);
  u16*  whi    = (u16*)w;    w += (size_t)HEADS * HDMAX * HID * sizeof(u16);
  u16*  wlo    = (u16*)w;    w += (size_t)HEADS * HDMAX * HID * sizeof(u16);
  float* a1    = (float*)w;  w += (size_t)N_NODES * HEADS * sizeof(float);
  float* a2    = (float*)w;  w += (size_t)N_NODES * HEADS * sizeof(float);
  float* denom = (float*)w;  w += (size_t)N_NODES * HEADS * sizeof(float);
  unsigned* smax = (unsigned*)w;

  // layer 0: [N,512] -> [N, 8*128]
  run_layer(features, D_IN, W0, b0, al0, alb0, ar0, arb0, HEADS, HID,
            src, dst, xhi, xlo, whi, wlo,
            ft, a1, a2, smax, denom, accum, xbuf, stream);
  // layer 1: [N,1024] -> [N, 8*128]  (out aliases input; stream order makes it safe)
  run_layer(xbuf, HEADS * HID, W1, b1, al1, alb1, ar1, arb1, HEADS, HID,
            src, dst, xhi, xlo, whi, wlo,
            ft, a1, a2, smax, denom, accum, xbuf, stream);
  // final: [N,1024] -> [N,64], single head
  run_layer(xbuf, HEADS * HID, Wf, bf_, alf, albf, arf, arbf, 1, C_OUT,
            src, dst, xhi, xlo, whi, wlo,
            ft, a1, a2, smax, denom, accum, (float*)d_out, stream);
}
